// MultiHeadLatentAttention_45268955300110
// MI455X (gfx1250) — compile-verified, matrix operand order checked by reading, not run on hardware
//
#include <hip/hip_runtime.h>
#include <hip/hip_bf16.h>

typedef __bf16 bf16;
typedef __attribute__((ext_vector_type(16))) __bf16 v16bf;
typedef __attribute__((ext_vector_type(2)))  __bf16 v2bf;
typedef __attribute__((ext_vector_type(8)))  float  v8f;
typedef __attribute__((ext_vector_type(4)))  unsigned int u32x4;
typedef __attribute__((ext_vector_type(8)))  int i32x8;
typedef __attribute__((ext_vector_type(4)))  int i32x4;

#define DM   1024
#define NH   16
#define DH   64
#define DL   256
#define DR   64
#define BB   2
#define SS   2048
#define MROWS (BB*SS)          // 4096 token rows

#if defined(__has_builtin)
#  if __has_builtin(__builtin_amdgcn_tensor_load_to_lds)
#    define HAVE_TDM 1
#  endif
#endif
#ifndef HAVE_TDM
#  define HAVE_TDM 0
#endif

// ---------------------------------------------------------------------------
// Elementwise conversions
// ---------------------------------------------------------------------------
__global__ void cvt_f32_to_bf16(const float* __restrict__ src,
                                bf16* __restrict__ dst, int n) {
  int i = blockIdx.x * 256 + threadIdx.x;
  if (i < n) dst[i] = (bf16)src[i];
}

// WT[n*K + k] = bf16( W[k*N + n] )   (transpose so GEMM B-loads are contiguous in K)
__global__ void wcvt_transpose(const float* __restrict__ W,
                               bf16* __restrict__ WT, int K, int N) {
  int idx = blockIdx.x * 256 + threadIdx.x;
  if (idx >= K * N) return;
  int n = idx / K, k = idx - n * K;
  WT[idx] = (bf16)W[(size_t)k * N + n];
}

// ---------------------------------------------------------------------------
// bf16 WMMA GEMM with 2x2 register blocking: each wave computes a 32x32 tile
// (4 x v_wmma_f32_16x16x32_bf16 per K-step, 4 fragments loaded -> 1:1 ratio).
// C[M,N] f32 = A[M,K] bf16 (row-major) x WT[N,K] bf16.
// ---------------------------------------------------------------------------
__global__ __launch_bounds__(256)
void gemm_bf16_nt(const bf16* __restrict__ A, const bf16* __restrict__ BT,
                  float* __restrict__ C, int M, int N, int K) {
  const int wid  = blockIdx.x * (blockDim.x >> 5) + (threadIdx.x >> 5);
  const int lane = threadIdx.x & 31;
  const int ntN  = N >> 5;
  const int ntiles = (M >> 5) * ntN;
  if (wid >= ntiles) return;
  const int tn = (wid % ntN) << 5;
  const int tm = (wid / ntN) << 5;
  const int r = lane & 15, half = lane >> 4;

  const bf16* a0p = A  + (size_t)(tm + r) * K;
  const bf16* a1p = A  + (size_t)(tm + 16 + r) * K;
  const bf16* b0p = BT + (size_t)(tn + r) * K;
  const bf16* b1p = BT + (size_t)(tn + 16 + r) * K;

  v8f c00 = {}, c01 = {}, c10 = {}, c11 = {};
  for (int kk = 0; kk < K; kk += 32) {
    __builtin_prefetch(a0p + kk + 64, 0, 1);
    __builtin_prefetch(a1p + kk + 64, 0, 1);
    __builtin_prefetch(b0p + kk + 64, 0, 1);
    __builtin_prefetch(b1p + kk + 64, 0, 1);
    v16bf a0, a1, b0, b1;
#pragma unroll
    for (int j = 0; j < 8; ++j) {
      const int k0 = kk + ((j >> 2) << 4) + (half << 3) + ((j & 3) << 1);
      v2bf x0 = *(const v2bf*)(a0p + k0);
      v2bf x1 = *(const v2bf*)(a1p + k0);
      v2bf y0 = *(const v2bf*)(b0p + k0);
      v2bf y1 = *(const v2bf*)(b1p + k0);
      a0[2 * j] = x0.x; a0[2 * j + 1] = x0.y;
      a1[2 * j] = x1.x; a1[2 * j + 1] = x1.y;
      b0[2 * j] = y0.x; b0[2 * j + 1] = y0.y;
      b1[2 * j] = y1.x; b1[2 * j + 1] = y1.y;
    }
    c00 = __builtin_amdgcn_wmma_f32_16x16x32_bf16(false, a0, false, b0, (short)0, c00, false, false);
    c01 = __builtin_amdgcn_wmma_f32_16x16x32_bf16(false, a0, false, b1, (short)0, c01, false, false);
    c10 = __builtin_amdgcn_wmma_f32_16x16x32_bf16(false, a1, false, b0, (short)0, c10, false, false);
    c11 = __builtin_amdgcn_wmma_f32_16x16x32_bf16(false, a1, false, b1, (short)0, c11, false, false);
  }
#pragma unroll
  for (int i = 0; i < 8; ++i) {
    const int m = (half << 3) + i;
    C[(size_t)(tm + m) * N + tn + r]           = c00[i];
    C[(size_t)(tm + m) * N + tn + 16 + r]      = c01[i];
    C[(size_t)(tm + 16 + m) * N + tn + r]      = c10[i];
    C[(size_t)(tm + 16 + m) * N + tn + 16 + r] = c11[i];
  }
}

// ---------------------------------------------------------------------------
// RoPE + pack: build per-head bf16 [q_c|q_r] (K=128), [k_c|k_r] (K=128), v (64),
// and write the f32 k_r output. One thread per (b,s,h,pair i<32).
// ---------------------------------------------------------------------------
__global__ void rope_pack(const float* __restrict__ qc, const float* __restrict__ qr,
                          const float* __restrict__ kc, const float* __restrict__ kr,
                          const float* __restrict__ vc,
                          bf16* __restrict__ qcat, bf16* __restrict__ kcat,
                          bf16* __restrict__ vb,   float* __restrict__ krout) {
  int idx = blockIdx.x * 256 + threadIdx.x;        // ((b*S+s)*NH+h)*32+i
  if (idx >= BB * SS * NH * 32) return;
  const int i = idx & 31;
  const int h = (idx >> 5) & (NH - 1);
  const int s = (idx >> 9) & (SS - 1);
  const int b = idx >> 20;

  const size_t src  = ((size_t)(b * SS + s)) * DM + h * DH;   // token-major f32
  const size_t hrow = ((size_t)(b * NH + h)) * SS + s;        // head-major

  const float theta = __expf(-(float)(2 * i) * (9.210340371976184f / 64.0f));
  const float ang = (float)s * theta;
  const float cs = __cosf(ang), sn = __sinf(ang);

  const float q1 = qr[src + 2 * i], q2 = qr[src + 2 * i + 1];
  const float k1 = kr[src + 2 * i], k2 = kr[src + 2 * i + 1];
  const float qo1 = q1 * cs - q2 * sn, qo2 = q1 * sn + q2 * cs;
  const float ko1 = k1 * cs - k2 * sn, ko2 = k1 * sn + k2 * cs;

  bf16* qd = qcat + hrow * 128;
  bf16* kd = kcat + hrow * 128;
  qd[2 * i]          = (bf16)qc[src + 2 * i];
  qd[2 * i + 1]      = (bf16)qc[src + 2 * i + 1];
  qd[64 + 2 * i]     = (bf16)qo1;
  qd[64 + 2 * i + 1] = (bf16)qo2;
  kd[2 * i]          = (bf16)kc[src + 2 * i];
  kd[2 * i + 1]      = (bf16)kc[src + 2 * i + 1];
  kd[64 + 2 * i]     = (bf16)ko1;
  kd[64 + 2 * i + 1] = (bf16)ko2;

  vb[hrow * 64 + 2 * i]     = (bf16)vc[src + 2 * i];
  vb[hrow * 64 + 2 * i + 1] = (bf16)vc[src + 2 * i + 1];

  krout[hrow * 64 + 2 * i]     = ko1;     // k_r output (B,NH,S,64) f32
  krout[hrow * 64 + 2 * i + 1] = ko2;
}

// ---------------------------------------------------------------------------
// TDM: 2-D contiguous tile (rows x cols bf16 elements) global -> LDS.
// D# built per CDNA5 ISA (group0/group1 bitfields).
// This toolchain exposes the 6-arg builtin (g0, g1, g2, g3, g1-ext, cpol).
// ---------------------------------------------------------------------------
#if HAVE_TDM
__device__ __forceinline__ void tdm_load_2d(const void* gaddr, unsigned lds_off,
                                            unsigned rows, unsigned cols,
                                            unsigned row_stride) {
  const unsigned long long ga = (unsigned long long)gaddr;
  u32x4 g0;
  g0[0] = 1u;                                            // count=1 (valid user D#)
  g0[1] = lds_off;                                       // lds_addr (bytes)
  g0[2] = (unsigned)(ga & 0xFFFFFFFFu);                  // global_addr[31:0]
  g0[3] = (unsigned)((ga >> 32) & 0x01FFFFFFu) | (2u << 30); // addr[56:32] | type=2
  i32x8 g1;
  g1[0] = (int)(1u << 16);                               // data_size=1 (2B), mask=0
  g1[1] = (int)((cols & 0xFFFFu) << 16);                 // tensor_dim0 lo16
  g1[2] = (int)(((cols >> 16) & 0xFFFFu) | ((rows & 0xFFFFu) << 16)); // dim0 hi | dim1 lo
  g1[3] = (int)(((rows >> 16) & 0xFFFFu) | ((cols & 0xFFFFu) << 16)); // dim1 hi | tile_dim0
  g1[4] = (int)(rows & 0xFFFFu);                         // tile_dim1 (tile_dim2=0)
  g1[5] = (int)row_stride;                               // tensor_dim0_stride lo32
  g1[6] = 0;                                             // stride0 hi | stride1 lo
  g1[7] = 0;                                             // stride1 hi
  const i32x4 z4 = {0, 0, 0, 0};
  const i32x8 z8 = {0, 0, 0, 0, 0, 0, 0, 0};
  __builtin_amdgcn_tensor_load_to_lds(g0, g1, z4, z4, z8, 0);
}
#endif

// ---------------------------------------------------------------------------
// Flash-style causal attention. One wave = one 16-query tile of one (b,h).
// Key step = 32 keys staged into per-wave LDS via the Tensor Data Mover:
// 8 score WMMAs (fused K=128 qc|qr) + 4 P.V WMMAs per step.
// ---------------------------------------------------------------------------
__global__ __launch_bounds__(128)
void mla_attn(const bf16* __restrict__ qcat, const bf16* __restrict__ kcat,
              const bf16* __restrict__ vb, bf16* __restrict__ ctx) {
  __shared__ bf16 kl[4][32 * 128];    // K tile per wave  (8 KB each)
  __shared__ bf16 vl[4][32 * 64];     // V tile per wave  (4 KB each)
  __shared__ bf16 pl[4][16 * 32];     // P staging per wave (1 KB each)

  const int wslot = threadIdx.x >> 5;
  const int lane  = threadIdx.x & 31;
  const int wid   = blockIdx.x * 4 + wslot;
  const int qt = wid & 127;                 // S/16 = 128 query tiles
  const int h  = (wid >> 7) & (NH - 1);
  const int b  = wid >> 11;
  const size_t bh = (size_t)(b * NH + h);
  const int r = lane & 15, half = lane >> 4;
  const int qbase = qt * 16;

  bf16* klw = kl[wslot];
  bf16* vlw = vl[wslot];
  bf16* my  = pl[wslot];
#if HAVE_TDM
  const unsigned koff = (unsigned)(uintptr_t)klw;
  const unsigned voff = (unsigned)(uintptr_t)vlw;
#endif

  // Q A-fragments, K = 128 split into 4 chunks of 32
  v16bf qf[4];
  const bf16* qrow = qcat + (bh * SS + qbase + r) * 128;
#pragma unroll
  for (int c = 0; c < 4; ++c)
#pragma unroll
    for (int j = 0; j < 8; ++j) {
      const int f = c * 32 + ((j >> 2) << 4) + (half << 3) + ((j & 3) << 1);
      v2bf p = *(const v2bf*)(qrow + f);
      qf[c][2 * j] = p.x; qf[c][2 * j + 1] = p.y;
    }

  v8f oacc[4] = {{}, {}, {}, {}};
  float rmax[8], rsum[8];
#pragma unroll
  for (int i = 0; i < 8; ++i) { rmax[i] = -1e30f; rsum[i] = 0.f; }

  const int nsteps = qt / 2 + 1;            // covers keys up to the diagonal
  for (int ks = 0; ks < nsteps; ++ks) {
    const int t0 = ks * 32;
    const bf16* kg = kcat + (bh * SS + t0) * 128;
    const bf16* vg = vb + (bh * SS + t0) * 64;

    // ---- stage K (32x128) and V (32x64) tiles into this wave's LDS ----
#if HAVE_TDM
    tdm_load_2d(kg, koff, 32, 128, 128);
    tdm_load_2d(vg, voff, 32, 64, 64);
    __builtin_amdgcn_s_wait_tensorcnt(0);
    asm volatile("" ::: "memory");
#else
    {
      const uint4* ks128 = (const uint4*)kg;
      uint4* kd128 = (uint4*)klw;
      for (int t = lane; t < (32 * 128 * 2) / 16; t += 32) kd128[t] = ks128[t];
      const uint4* vs128 = (const uint4*)vg;
      uint4* vd128 = (uint4*)vlw;
      for (int t = lane; t < (32 * 64 * 2) / 16; t += 32) vd128[t] = vs128[t];
      asm volatile("" ::: "memory");
    }
#endif

    // ---- scores S[16 x 32] = ([qc|qr] . [kc|kr]^T) from LDS ----
    v8f s0a = {}, s1a = {};
    const bf16* k0row = klw + (size_t)r * 128;          // key t0 + r
    const bf16* k1row = klw + (size_t)(16 + r) * 128;   // key t0 + 16 + r
#pragma unroll
    for (int c = 0; c < 4; ++c) {
      v16bf kb0, kb1;
#pragma unroll
      for (int j = 0; j < 8; ++j) {
        const int f = c * 32 + ((j >> 2) << 4) + (half << 3) + ((j & 3) << 1);
        v2bf p0 = *(const v2bf*)(k0row + f);
        v2bf p1 = *(const v2bf*)(k1row + f);
        kb0[2 * j] = p0.x; kb0[2 * j + 1] = p0.y;
        kb1[2 * j] = p1.x; kb1[2 * j + 1] = p1.y;
      }
      s0a = __builtin_amdgcn_wmma_f32_16x16x32_bf16(false, qf[c], false, kb0,
                                                    (short)0, s0a, false, false);
      s1a = __builtin_amdgcn_wmma_f32_16x16x32_bf16(false, qf[c], false, kb1,
                                                    (short)0, s1a, false, false);
    }

    // ---- online softmax update (per row; rows live across half-waves) ----
#pragma unroll
    for (int i = 0; i < 8; ++i) {
      const int m  = (half << 3) + i;
      const int sg = qbase + m;
      const int ta = t0 + r;
      float sa = s0a[i] * 0.125f + ((ta      <= sg) ? 0.f : -1e9f);
      float sb = s1a[i] * 0.125f + ((ta + 16 <= sg) ? 0.f : -1e9f);
      float tmax = fmaxf(sa, sb);
#pragma unroll
      for (int d = 1; d < 16; d <<= 1)
        tmax = fmaxf(tmax, __shfl_xor(tmax, d, 16));
      const float nmax = fmaxf(rmax[i], tmax);
      const float corr = __expf(rmax[i] - nmax);
      rmax[i] = nmax;
      const float pa = __expf(sa - nmax);
      const float pb = __expf(sb - nmax);
      float ps = pa + pb;
#pragma unroll
      for (int d = 1; d < 16; d <<= 1)
        ps += __shfl_xor(ps, d, 16);
      rsum[i] = rsum[i] * corr + ps;
#pragma unroll
      for (int g = 0; g < 4; ++g) oacc[g][i] *= corr;
      my[m * 32 + r]      = (bf16)pa;       // stage P for A-layout reload
      my[m * 32 + 16 + r] = (bf16)pb;
    }

    // ---- reload P as A-fragment (16x32), same-wave LDS RAW ----
    v16bf pf;
#pragma unroll
    for (int j = 0; j < 8; ++j) {
      const int k0 = ((j >> 2) << 4) + (half << 3) + ((j & 3) << 1);
      v2bf p = *(const v2bf*)(my + r * 32 + k0);
      pf[2 * j] = p.x; pf[2 * j + 1] = p.y;
    }

    // ---- O[16x64] += P[16x32] . V[32x64]  (V from LDS) ----
#pragma unroll
    for (int g = 0; g < 4; ++g) {
      v16bf vf;
#pragma unroll
      for (int j = 0; j < 8; ++j) {
        const int k0 = ((j >> 2) << 4) + (half << 3) + ((j & 3) << 1);
        vf[2 * j]     = vlw[(size_t)k0 * 64 + g * 16 + r];
        vf[2 * j + 1] = vlw[(size_t)(k0 + 1) * 64 + g * 16 + r];
      }
      oacc[g] = __builtin_amdgcn_wmma_f32_16x16x32_bf16(false, pf, false, vf,
                                                        (short)0, oacc[g], false, false);
    }
  }

  // ---- normalize + store ctx bf16 (token-major, head columns) ----
#pragma unroll
  for (int i = 0; i < 8; ++i) {
    const float inv = 1.0f / rsum[i];
    const int m = (half << 3) + i;
    bf16* crow = ctx + ((size_t)(b * SS + qbase + m)) * DM + h * DH;
#pragma unroll
    for (int g = 0; g < 4; ++g)
      crow[g * 16 + r] = (bf16)(oacc[g][i] * inv);
  }
}

// ---------------------------------------------------------------------------
// Host-side launch
// ---------------------------------------------------------------------------
extern "C" void kernel_launch(void* const* d_in, const int* in_sizes, int n_in,
                              void* d_out, int out_size, void* d_ws, size_t ws_size,
                              hipStream_t stream) {
  const float* x      = (const float*)d_in[0];
  // d_in[1] = attn_mask (causal; applied analytically)
  const float* W_q    = (const float*)d_in[2];
  const float* W_dkv  = (const float*)d_in[3];
  const float* W_uk   = (const float*)d_in[4];
  const float* W_uv   = (const float*)d_in[5];
  const float* W_kpos = (const float*)d_in[6];
  const float* W_qpos = (const float*)d_in[7];
  const float* W_o    = (const float*)d_in[8];

  float* out_main = (float*)d_out;                         // (B,S,D)      4,194,304
  float* out_ckv  = out_main + (size_t)MROWS * DM;         // (B,S,256)    1,048,576
  float* out_kr   = out_ckv + (size_t)MROWS * DL;          // (B,NH,S,64)  4,194,304

  char* wsb = (char*)d_ws;
  size_t off = 0;
  auto alloc = [&](size_t bytes) -> void* {
    off = (off + 255) & ~(size_t)255;
    void* p = wsb + off;
    off += bytes;
    return p;
  };
  bf16* xb      = (bf16*)alloc((size_t)MROWS * DM * 2);
  bf16* Wq_t    = (bf16*)alloc((size_t)DM * DM * 2);
  bf16* Wqpos_t = (bf16*)alloc((size_t)DM * DM * 2);
  bf16* Wkpos_t = (bf16*)alloc((size_t)DM * DM * 2);
  bf16* Wo_t    = (bf16*)alloc((size_t)DM * DM * 2);
  bf16* Wdkv_t  = (bf16*)alloc((size_t)DM * DL * 2);
  bf16* Wuk_t   = (bf16*)alloc((size_t)DL * DM * 2);
  bf16* Wuv_t   = (bf16*)alloc((size_t)DL * DM * 2);
  float* qc  = (float*)alloc((size_t)MROWS * DM * 4);
  float* qr  = (float*)alloc((size_t)MROWS * DM * 4);
  float* krr = (float*)alloc((size_t)MROWS * DM * 4);
  float* kc  = (float*)alloc((size_t)MROWS * DM * 4);
  float* vc  = (float*)alloc((size_t)MROWS * DM * 4);
  bf16* ckvb = (bf16*)alloc((size_t)MROWS * DL * 2);
  bf16* qcat = (bf16*)alloc((size_t)BB * NH * SS * 128 * 2);
  bf16* kcat = (bf16*)alloc((size_t)BB * NH * SS * 128 * 2);
  bf16* vbuf = (bf16*)alloc((size_t)BB * NH * SS * 64 * 2);
  bf16* ctxb = (bf16*)alloc((size_t)MROWS * DM * 2);

  auto blocks_of = [](size_t n) { return (unsigned)((n + 255) / 256); };
  auto gemm_grid = [](int M, int N) { return (unsigned)(((M >> 5) * (N >> 5) + 7) / 8); };

  // 1) precision conversion
  cvt_f32_to_bf16<<<blocks_of((size_t)MROWS * DM), 256, 0, stream>>>(x, xb, MROWS * DM);
  wcvt_transpose<<<blocks_of((size_t)DM * DM), 256, 0, stream>>>(W_q,    Wq_t,    DM, DM);
  wcvt_transpose<<<blocks_of((size_t)DM * DM), 256, 0, stream>>>(W_qpos, Wqpos_t, DM, DM);
  wcvt_transpose<<<blocks_of((size_t)DM * DM), 256, 0, stream>>>(W_kpos, Wkpos_t, DM, DM);
  wcvt_transpose<<<blocks_of((size_t)DM * DM), 256, 0, stream>>>(W_o,    Wo_t,    DM, DM);
  wcvt_transpose<<<blocks_of((size_t)DM * DL), 256, 0, stream>>>(W_dkv,  Wdkv_t,  DM, DL);
  wcvt_transpose<<<blocks_of((size_t)DL * DM), 256, 0, stream>>>(W_uk,   Wuk_t,   DL, DM);
  wcvt_transpose<<<blocks_of((size_t)DL * DM), 256, 0, stream>>>(W_uv,   Wuv_t,   DL, DM);

  // 2) projection GEMMs
  gemm_bf16_nt<<<gemm_grid(MROWS, DM), 256, 0, stream>>>(xb, Wq_t,    qc,      MROWS, DM, DM);
  gemm_bf16_nt<<<gemm_grid(MROWS, DL), 256, 0, stream>>>(xb, Wdkv_t,  out_ckv, MROWS, DL, DM);
  cvt_f32_to_bf16<<<blocks_of((size_t)MROWS * DL), 256, 0, stream>>>(out_ckv, ckvb, MROWS * DL);
  gemm_bf16_nt<<<gemm_grid(MROWS, DM), 256, 0, stream>>>(xb,   Wqpos_t, qr,  MROWS, DM, DM);
  gemm_bf16_nt<<<gemm_grid(MROWS, DM), 256, 0, stream>>>(xb,   Wkpos_t, krr, MROWS, DM, DM);
  gemm_bf16_nt<<<gemm_grid(MROWS, DM), 256, 0, stream>>>(ckvb, Wuk_t,   kc,  MROWS, DM, DL);
  gemm_bf16_nt<<<gemm_grid(MROWS, DM), 256, 0, stream>>>(ckvb, Wuv_t,   vc,  MROWS, DM, DL);

  // 3) RoPE + head-major packing (+ k_r output)
  rope_pack<<<blocks_of((size_t)BB * SS * NH * 32), 256, 0, stream>>>(
      qc, qr, kc, krr, vc, qcat, kcat, vbuf, out_kr);

  // 4) causal flash attention (B*NH*(S/16) waves / 4 per block)
  mla_attn<<<(unsigned)(BB * NH * (SS / 16) / 4), 128, 0, stream>>>(qcat, kcat, vbuf, ctxb);

  // 5) output projection
  gemm_bf16_nt<<<gemm_grid(MROWS, DM), 256, 0, stream>>>(ctxb, Wo_t, out_main, MROWS, DM, DM);
}
